// GAT_2946347565081
// MI455X (gfx1250) — matrix operand order repaired and checked by reading, not compiled, not run on hardware
//
#include <hip/hip_runtime.h>
#include <hip/hip_bf16.h>

// ---------------------------------------------------------------------------
// GAT (2-layer, HID=16, 16 classes) for MI455X gfx1250.
//  - x@W1 and the 16x16 GEMMs run on v_wmma_f32_16x16x32_bf16 (f32 accum).
//  - alpha_s/alpha_d folded into a second WMMA accumulator (B2=[W a_s, W a_d]).
//  - B fragments fragment-major in LDS -> 2x ds_load_b128 per chunk.
//  - Edge aggregation: 16 threads/edge so each atomic instruction touches only
//    2 contiguous 64B rows (L2-atomic friendly); ordered-uint float atomicMax.
//  - Softmax reductions via DPP ladder (pure VALU, no LDS/bpermute traffic).
// ---------------------------------------------------------------------------

#define HID 16
#define NEG_SLOPE 0.2f
#define BLK 256

typedef __attribute__((ext_vector_type(16))) __bf16 v16bf;
typedef __attribute__((ext_vector_type(8)))  float  v8f;

static __device__ __forceinline__ __bf16 f32_to_bf16(float f) {
  return (__bf16)f;                         // native v_cvt(_pk)_bf16_f32, RNE
}
static __device__ __forceinline__ unsigned short bf16_bits_from_f32(float f) {
  __bf16 h = (__bf16)f;
  unsigned short u; __builtin_memcpy(&u, &h, 2); return u;
}

// Order-preserving float <-> uint encoding for atomicMax on floats.
static __device__ __forceinline__ unsigned fenc(float x) {
  unsigned u = __float_as_uint(x);
  return (u & 0x80000000u) ? ~u : (u | 0x80000000u);
}
static __device__ __forceinline__ float fdec(unsigned e) {
  unsigned u = (e & 0x80000000u) ? (e ^ 0x80000000u) : ~e;
  return __uint_as_float(u);
}
#define ENC_NEG_INF 0x007FFFFFu   // fenc(-inf)

// ---------------------------------------------------------------------------
// DPP 16-lane (row) reductions: xor1, xor2, then mirrors (value-equivalent to
// xor4/xor8 once subgroups are uniform).  Pure VALU, no LDS.
// dpp_ctrl must be an ICE -> template parameter.
// ---------------------------------------------------------------------------
#define DPP_QUAD_PERM_1032  0xB1
#define DPP_QUAD_PERM_2301  0x4E
#define DPP_ROW_HALF_MIRROR 0x141
#define DPP_ROW_MIRROR      0x140

template<int CTRL>
static __device__ __forceinline__ float dpp_mov_f32(float v) {
  return __int_as_float(__builtin_amdgcn_update_dpp(
      0, __float_as_int(v), CTRL, 0xF, 0xF, true));
}
static __device__ __forceinline__ float dpp16_sum(float v) {
  v += dpp_mov_f32<DPP_QUAD_PERM_1032>(v);
  v += dpp_mov_f32<DPP_QUAD_PERM_2301>(v);
  v += dpp_mov_f32<DPP_ROW_HALF_MIRROR>(v);
  v += dpp_mov_f32<DPP_ROW_MIRROR>(v);
  return v;
}
static __device__ __forceinline__ float dpp16_max(float v) {
  v = fmaxf(v, dpp_mov_f32<DPP_QUAD_PERM_1032>(v));
  v = fmaxf(v, dpp_mov_f32<DPP_QUAD_PERM_2301>(v));
  v = fmaxf(v, dpp_mov_f32<DPP_ROW_HALF_MIRROR>(v));
  v = fmaxf(v, dpp_mov_f32<DPP_ROW_MIRROR>(v));
  return v;
}

// ---------------------------------------------------------------------------
// WMMA fragment helpers (layouts per CDNA5 ISA 7.12.2, wave32)
// ---------------------------------------------------------------------------

// A fragment: 16x32 bf16.  lanes 0-15: row M=lane,    K = kb+0..7 & kb+16..23
//                          lanes 16-31: row M=lane-16, K = kb+8..15 & kb+24..31
template<int KDIM>
static __device__ __forceinline__ v16bf load_a_frag(const float* __restrict__ xrow,
                                                    int kb, int hi) {
  v16bf a;
  const int k0 = kb + hi * 8;
#pragma unroll
  for (int j = 0; j < 8; ++j) {
    const int k1 = k0 + j;
    const int k2 = k0 + 16 + j;
    a[j]     = f32_to_bf16((k1 < KDIM) ? xrow[k1] : 0.0f);
    a[8 + j] = f32_to_bf16((k2 < KDIM) ? xrow[k2] : 0.0f);
  }
  return a;
}

// Fragment-major LDS layout: sB[ch][lane][j], 16 contiguous ushort per lane.
// Lane n holds column (n&15), K = ch*32 + (j>>3)*16 + (n>>4)*8 + (j&7).
static __device__ __forceinline__ v16bf load_b_frag_packed(
    const unsigned short* __restrict__ sB, int ch, int lane) {
  v16bf b;
  __builtin_memcpy(&b, sB + (ch << 9) + (lane << 4), 32);  // 2x ds_load_b128
  return b;
}

// Fill fragment-major B from row-major W (KDIM x 16), zero-padded to KPAD.
template<int KDIM>
static __device__ __forceinline__ void fill_b_frags(unsigned short* sB,
                                                    const float* __restrict__ W,
                                                    int t) {
  constexpr int KPAD = (KDIM < 32) ? 32 : KDIM;
  constexpr int TOT  = (KPAD / 32) * 512;
  for (int i = t; i < TOT; i += BLK) {
    const int ch = i >> 9, r = i & 511, lane = r >> 4, j = r & 15;
    const int k = ch * 32 + ((j >> 3) << 4) + ((lane >> 4) << 3) + (j & 7);
    const int col = lane & 15;
    sB[i] = (k < KDIM) ? bf16_bits_from_f32(W[k * HID + col]) : (unsigned short)0;
  }
}

// Fill fragment-major B2: col0 = w_as[k], col1 = w_ad[k], other cols 0.
template<int KDIM>
static __device__ __forceinline__ void fill_b2_frags(unsigned short* sB2,
                                                     const float* __restrict__ w_as,
                                                     const float* __restrict__ w_ad,
                                                     int t) {
  constexpr int KPAD = (KDIM < 32) ? 32 : KDIM;
  constexpr int TOT  = (KPAD / 32) * 512;
  for (int i = t; i < TOT; i += BLK) {
    const int ch = i >> 9, r = i & 511, lane = r >> 4, j = r & 15;
    const int k = ch * 32 + ((j >> 3) << 4) + ((lane >> 4) << 3) + (j & 7);
    const int col = lane & 15;
    float v = 0.0f;
    if (k < KDIM) {
      if (col == 0) v = w_as[k];
      else if (col == 1) v = w_ad[k];
    }
    sB2[i] = bf16_bits_from_f32(v);
  }
}

// ---------------------------------------------------------------------------
// Kernels
// ---------------------------------------------------------------------------

// w_as = W @ a_s ; w_ad = W @ a_d   (W is K x 16 row-major)
__global__ void gat_prep_kernel(const float* __restrict__ W,
                                const float* __restrict__ a_s,
                                const float* __restrict__ a_d,
                                float* __restrict__ w_as,
                                float* __restrict__ w_ad, int K) {
  const int k = blockIdx.x * blockDim.x + threadIdx.x;
  if (k >= K) return;
  float s = 0.0f, d = 0.0f;
#pragma unroll
  for (int c = 0; c < HID; ++c) {
    const float w = W[k * HID + c];
    s += w * a_s[c];
    d += w * a_d[c];
  }
  w_as[k] = s;
  w_ad[k] = d;
}

// Re-init per call (d_ws is poisoned): m=enc(-inf), z=0, acc[16N]=0
__global__ void gat_init_kernel(unsigned* __restrict__ m_enc, float* __restrict__ z,
                                float* __restrict__ acc, int n) {
  const int i = blockIdx.x * blockDim.x + threadIdx.x;
  if (i < 16 * n) acc[i] = 0.0f;
  if (i < n) { m_enc[i] = ENC_NEG_INF; z[i] = 0.0f; }
}

// H = X @ W, AS = X @ (W a_s), AD = X @ (W a_d) -- dual-accumulator WMMA.
template<int KDIM>
__global__ void gat_gemm_alpha_kernel(const float* __restrict__ X,
                                      const float* __restrict__ W,
                                      const float* __restrict__ w_as,
                                      const float* __restrict__ w_ad,
                                      float* __restrict__ H,
                                      float* __restrict__ AS,
                                      float* __restrict__ AD,
                                      int numTiles) {
  constexpr int KPAD   = (KDIM < 32) ? 32 : KDIM;
  constexpr int CHUNKS = KPAD / 32;
  __shared__ alignas(32) unsigned short sB [CHUNKS * 512];
  __shared__ alignas(32) unsigned short sB2[CHUNKS * 512];
  const int t = threadIdx.x;
  fill_b_frags<KDIM>(sB, W, t);
  fill_b2_frags<KDIM>(sB2, w_as, w_ad, t);
  __syncthreads();

  const int wave = t >> 5, lane = t & 31;
  const int tile = blockIdx.x * (BLK >> 5) + wave;
  if (tile >= numTiles) return;           // wave-uniform: EXEC stays all-ones
  const int rowBase = tile * 16;

  const int l = lane & 15, hi = lane >> 4;
  const float* xrow = X + (size_t)(rowBase + l) * KDIM;
  v8f c  = {};
  v8f c2 = {};
#pragma unroll
  for (int ch = 0; ch < CHUNKS; ++ch) {
    v16bf a  = load_a_frag<KDIM>(xrow, ch * 32, hi);
    v16bf b  = load_b_frag_packed(sB,  ch, lane);
    v16bf b2 = load_b_frag_packed(sB2, ch, lane);
    c  = __builtin_amdgcn_wmma_f32_16x16x32_bf16(false, a, false, b,  (short)0, c,
                                                 false, false);
    c2 = __builtin_amdgcn_wmma_f32_16x16x32_bf16(false, a, false, b2, (short)0, c2,
                                                 false, false);
  }

  // C layout: VGPR r -> rows rowBase+r (lanes 0-15) / rowBase+r+8 (lanes 16-31)
#pragma unroll
  for (int r = 0; r < 8; ++r)
    H[(rowBase + r + hi * 8) * HID + l] = c[r];
  if (l < 2) {
    float* dst = (l == 0) ? AS : AD;      // alpha_s in col 0, alpha_d in col 1
#pragma unroll
    for (int r = 0; r < 8; ++r)
      dst[rowBase + r + hi * 8] = c2[r];
  }
}

// Pass A: segment max of leaky-relu logits per dst (self-loops appended).
__global__ void gat_edge_max_kernel(const int* __restrict__ ei, int E, int N,
                                    const float* __restrict__ AS,
                                    const float* __restrict__ AD,
                                    unsigned* __restrict__ m_enc) {
  const int e = blockIdx.x * blockDim.x + threadIdx.x;
  if (e >= E + N) return;
  int s, d;
  if (e < E) { s = ei[e]; d = ei[E + e]; } else { s = d = e - E; }
  float lg = AS[s] + AD[d];
  lg = (lg > 0.0f) ? lg : lg * NEG_SLOPE;
  atomicMax(&m_enc[d], fenc(lg));
}

// Pass B: 16 threads per edge (lane = channel) -> each 32-lane atomic covers
// just 2 contiguous 64B rows.  ev recompute is broadcast-L0 + cheap VALU.
__global__ void gat_edge_sum_kernel(const int* __restrict__ ei, int E, int N,
                                    const float* __restrict__ AS,
                                    const float* __restrict__ AD,
                                    const unsigned* __restrict__ m_enc,
                                    const float* __restrict__ H,
                                    float* __restrict__ z,
                                    float* __restrict__ acc) {
  const unsigned idx = blockIdx.x * blockDim.x + threadIdx.x;
  const int e = (int)(idx >> 4);
  const int c = (int)(idx & 15u);
  if (e >= E + N) return;
  int s, d;
  if (e < E) { s = ei[e]; d = ei[E + e]; } else { s = d = e - E; }
  float lg = AS[s] + AD[d];
  lg = (lg > 0.0f) ? lg : lg * NEG_SLOPE;
  const float ev = __expf(lg - fdec(m_enc[d]));
  if (c == 0) atomicAdd(&z[d], ev);
  atomicAdd(&acc[(size_t)d * HID + c], ev * H[(size_t)s * HID + c]);
}

// out[n,c] = relu(acc[n,c]/(z[n]+1e-16) + b[c])
__global__ void gat_finalize_kernel(const float* __restrict__ acc,
                                    const float* __restrict__ z,
                                    const float* __restrict__ bias,
                                    float* __restrict__ out, int n) {
  const int i = blockIdx.x * blockDim.x + threadIdx.x;
  if (i >= 16 * n) return;
  const int node = i >> 4, c = i & 15;
  const float v = acc[i] / (z[node] + 1e-16f) + bias[c];
  out[i] = fmaxf(v, 0.0f);
}

// OUT = softmax(X @ Wout + bout) rows; K=16 zero-padded WMMA + DPP softmax.
__global__ void gat_gemm_softmax_kernel(const float* __restrict__ X,
                                        const float* __restrict__ W,
                                        const float* __restrict__ bias,
                                        float* __restrict__ OUT,
                                        int numTiles) {
  constexpr int KDIM = 16;
  __shared__ alignas(32) unsigned short sB[512];
  const int t = threadIdx.x;
  fill_b_frags<KDIM>(sB, W, t);
  __syncthreads();

  const int wave = t >> 5, lane = t & 31;
  const int tile = blockIdx.x * (BLK >> 5) + wave;
  if (tile >= numTiles) return;
  const int rowBase = tile * 16;

  const int l = lane & 15, hi = lane >> 4;
  const float* xrow = X + (size_t)(rowBase + l) * KDIM;
  v16bf a = load_a_frag<KDIM>(xrow, 0, hi);
  v16bf b = load_b_frag_packed(sB, 0, lane);
  v8f c = {};
  c = __builtin_amdgcn_wmma_f32_16x16x32_bf16(false, a, false, b, (short)0, c,
                                              false, false);

  const float bv = bias[l];
#pragma unroll
  for (int r = 0; r < 8; ++r) {
    const int row = rowBase + r + hi * 8;
    const float v = c[r] + bv;
    const float mx = dpp16_max(v);
    const float ev = __expf(v - mx);
    const float sm = dpp16_sum(ev);
    OUT[row * HID + l] = ev / sm;
  }
}

// ---------------------------------------------------------------------------
// Launch
// ---------------------------------------------------------------------------
extern "C" void kernel_launch(void* const* d_in, const int* in_sizes, int n_in,
                              void* d_out, int out_size, void* d_ws, size_t ws_size,
                              hipStream_t stream) {
  const float* x    = (const float*)d_in[0];
  const float* W1   = (const float*)d_in[1];
  const float* as1w = (const float*)d_in[2];
  const float* ad1w = (const float*)d_in[3];
  const float* b1   = (const float*)d_in[4];
  const float* W2   = (const float*)d_in[5];
  const float* as2w = (const float*)d_in[6];
  const float* ad2w = (const float*)d_in[7];
  const float* b2   = (const float*)d_in[8];
  const float* Wout = (const float*)d_in[9];
  const float* bout = (const float*)d_in[10];
  const int*   ei   = (const int*)d_in[11];

  const int N = in_sizes[0] / 512;      // 100000 (multiple of 16)
  const int E = in_sizes[11] / 2;       // 3200000

  // Workspace layout (floats); float4-accessed buffers stay 16B aligned.
  float* ws   = (float*)d_ws;
  float*    h1    = ws;                       // 16N
  float*    as1   = h1  + (size_t)16 * N;     // N
  float*    ad1   = as1 + N;                  // N
  float*    as2   = ad1 + N;                  // N
  float*    ad2   = as2 + N;                  // N
  unsigned* m_enc = (unsigned*)(ad2 + N);     // N
  float*    z     = (float*)(m_enc + N);      // N
  float*    acc   = z + N;                    // 16N
  float*    h1r   = acc + (size_t)16 * N;     // 16N
  float*    h2    = h1r + (size_t)16 * N;     // 16N
  float*    h2r   = h2  + (size_t)16 * N;     // 16N
  float*    w_as1 = h2r + (size_t)16 * N;     // 512
  float*    w_ad1 = w_as1 + 512;              // 512
  float*    w_as2 = w_ad1 + 512;              // 16
  float*    w_ad2 = w_as2 + 16;               // 16
  float*    out   = (float*)d_out;            // 16N softmax probs

  const int numTiles = N / 16;
  const int gemmBlks = (numTiles + (BLK / 32) - 1) / (BLK / 32);
  const int nodeBlks = (16 * N + BLK - 1) / BLK;
  const int edgeBlks = (E + N + BLK - 1) / BLK;
  const int edge16Blks = (int)(((long long)(E + N) * 16 + BLK - 1) / BLK);

  // ---- Prep: fold attention vectors through the weights ----
  gat_prep_kernel<<<(512 + 63) / 64, 64, 0, stream>>>(W1, as1w, ad1w,
                                                      w_as1, w_ad1, 512);
  gat_prep_kernel<<<1, 64, 0, stream>>>(W2, as2w, ad2w, w_as2, w_ad2, 16);

  // ---- Layer 1 ----
  gat_init_kernel<<<nodeBlks, BLK, 0, stream>>>(m_enc, z, acc, N);
  gat_gemm_alpha_kernel<512><<<gemmBlks, BLK, 0, stream>>>(
      x, W1, w_as1, w_ad1, h1, as1, ad1, numTiles);
  gat_edge_max_kernel<<<edgeBlks, BLK, 0, stream>>>(ei, E, N, as1, ad1, m_enc);
  gat_edge_sum_kernel<<<edge16Blks, BLK, 0, stream>>>(ei, E, N, as1, ad1, m_enc,
                                                      h1, z, acc);
  gat_finalize_kernel<<<nodeBlks, BLK, 0, stream>>>(acc, z, b1, h1r, N);

  // ---- Layer 2 ----
  gat_init_kernel<<<nodeBlks, BLK, 0, stream>>>(m_enc, z, acc, N);
  gat_gemm_alpha_kernel<16><<<gemmBlks, BLK, 0, stream>>>(
      h1r, W2, w_as2, w_ad2, h2, as2, ad2, numTiles);
  gat_edge_max_kernel<<<edgeBlks, BLK, 0, stream>>>(ei, E, N, as2, ad2, m_enc);
  gat_edge_sum_kernel<<<edge16Blks, BLK, 0, stream>>>(ei, E, N, as2, ad2, m_enc,
                                                      h2, z, acc);
  gat_finalize_kernel<<<nodeBlks, BLK, 0, stream>>>(acc, z, b2, h2r, N);

  // ---- Output head ----
  gat_gemm_softmax_kernel<<<gemmBlks, BLK, 0, stream>>>(h2r, Wout, bout, out,
                                                        numTiles);
}
